// Model_72232759984535
// MI455X (gfx1250) — matrix operand order repaired and checked by reading, not compiled, not run on hardware
//
#include <hip/hip_runtime.h>
#include <hip/hip_bf16.h>

// ---------------------------------------------------------------------------
// CDNA5 (gfx1250) 6-layer graph autoencoder:
//   per layer:  Z = relu( A_sparse @ (X @ W + b) )  batched over B=256.
// Dense GEMMs: bf16 WMMA (v_wmma_f32_16x16x32_bf16), fp32 accumulate.
// Weights pre-packed (zero padded to 4-tile groups) into B-fragment layout
// [group][ks][t][lane][16] so the GEMM hot loop is branch-free: clause-
// batched b128 loads with immediate offsets + 4 WMMAs, EXEC all-1s.
// ReLU is a template parameter (no per-element cndmask in the hot loop).
// SpMM: float atomics into an L2-resident buffer.
// ---------------------------------------------------------------------------

typedef __attribute__((ext_vector_type(16))) __bf16 v16bf;
typedef __attribute__((ext_vector_type(8)))  float  v8f;
typedef __attribute__((ext_vector_type(4)))  float  v4f;

__device__ __forceinline__ __bf16 f2bf(float f) { return (__bf16)f; }

// ---------------------------------------------------------------------------
// Pack W[K,Nout] (fp32, row-major) -> Wp in grouped WMMA B-fragment order:
//   ng = nt>>2, t = nt&3
//   Wp[(((ng*ksteps + ks)*4 + t)*32 + lane)*16 + j]
//     = bf16(W[ks*32 + (j<8?j:j+8) + 8*(lane>>4)][nt*16 + (lane&15)])
// zero-padded for K tail and for nt in [ntn, ntn_pad).  Runs once per layer.
// ---------------------------------------------------------------------------
__global__ void pack_w(const float* __restrict__ W, __bf16* __restrict__ Wp,
                       int K, int Nout, int ntn_pad, int ksteps)
{
    const int idx = blockIdx.x * blockDim.x + threadIdx.x;
    const int totalLanes = ntn_pad * ksteps * 32;
    if (idx >= totalLanes) return;

    const int lane = idx & 31;
    const int rest = idx >> 5;
    const int ks   = rest % ksteps;
    const int nt   = rest / ksteps;
    const int ng   = nt >> 2;
    const int tt   = nt & 3;
    const int half = lane >> 4;
    const int col  = (nt << 4) + (lane & 15);

    __bf16* __restrict__ dst =
        Wp + ((((size_t)ng * ksteps + ks) * 4 + tt) * 32 + lane) * 16;
#pragma unroll
    for (int j = 0; j < 16; ++j) {
        const int kk = (ks << 5) + (j < 8 ? j : j + 8) + (half << 3);
        const float wv = (kk < K && col < Nout) ? W[(size_t)kk * Nout + col] : 0.0f;
        dst[j] = f2bf(wv);
    }
}

// ---------------------------------------------------------------------------
// GEMM + bias:  Y[M,Nout] = (RELU ? relu(X) : X)[M,K] @ W + b
// One wave32 computes a 16x64 strip (4 N-tiles share one A fragment).
// A (16x32 bf16): lane l -> row (l&15); elem j -> K = (j<8?j:j+8) + 8*(l>>4)
// C/D (16x16 f32): lane l -> col (l&15); elem v -> row v + 8*(l>>4)
// ---------------------------------------------------------------------------
template <bool RELU>
__global__ void gemm_bias_wmma(const float* __restrict__ X,
                               const __bf16* __restrict__ Wp,
                               const float* __restrict__ bias,
                               float* __restrict__ Y,
                               int M, int K, int Nout, int ntg, int ksteps)
{
    const int lane = threadIdx.x & 31;
    const int wave = threadIdx.x >> 5;
    const int total = (M >> 4) * ntg;            // M = 65280, multiple of 16

    const int wid = blockIdx.x * (blockDim.x >> 5) + wave;
    if (wid >= total) return;                    // uniform within a wave

    const int mt   = wid / ntg;
    const int ng   = wid - mt * ntg;
    const int row0 = mt << 4;
    const int half = lane >> 4;
    const int mrow = row0 + (lane & 15);

    const float*  __restrict__ xr = X + (size_t)mrow * K;
    // 4 sub-tiles * 32 lanes * 16 elems = 2048 bf16 per K-step for this group
    const __bf16* __restrict__ wg =
        Wp + (size_t)ng * ksteps * 2048 + (size_t)lane * 16;

    v8f acc[4] = {};
    const int kfull = K >> 5;                    // full 32-wide K steps

    int ks = 0;
    for (; ks < kfull; ++ks) {                   // hot loop: branch-free body
        const int k0 = ks << 5;
        const float* p0 = xr + k0 + (half << 3);
        v4f x0 = *(const v4f*)(p0);
        v4f x1 = *(const v4f*)(p0 + 4);
        v4f x2 = *(const v4f*)(p0 + 16);
        v4f x3 = *(const v4f*)(p0 + 20);

        float xv[16];
#pragma unroll
        for (int j = 0; j < 4; ++j) {
            xv[j]      = x0[j];
            xv[4 + j]  = x1[j];
            xv[8 + j]  = x2[j];
            xv[12 + j] = x3[j];
        }
        v16bf a;
#pragma unroll
        for (int j = 0; j < 16; ++j) {
            float v = xv[j];
            if (RELU) v = fmaxf(v, 0.0f);        // compile-time: no cndmask
            a[j] = f2bf(v);
        }
        const __bf16* bp = wg + (size_t)ks * 2048;
#pragma unroll
        for (int t = 0; t < 4; ++t) {
            const v16bf b = *(const v16bf*)(bp + t * 512);
            acc[t] = __builtin_amdgcn_wmma_f32_16x16x32_bf16(
                         false, a, false, b, (short)0, acc[t], false, false);
        }
    }

    if (ks < ksteps) {                           // single K-tail step
        const int k0 = ks << 5;
        v16bf a;
#pragma unroll
        for (int j = 0; j < 16; ++j) {
            const int kk = k0 + (j < 8 ? j : j + 8) + (half << 3);
            const int kc = kk < K ? kk : K - 1;  // clamp: unconditional load
            float v = xr[kc];
            v = (kk < K) ? v : 0.0f;             // cndmask, no EXEC branch
            if (RELU) v = fmaxf(v, 0.0f);
            a[j] = f2bf(v);
        }
        const __bf16* bp = wg + (size_t)ks * 2048;
#pragma unroll
        for (int t = 0; t < 4; ++t) {
            const v16bf b = *(const v16bf*)(bp + t * 512);
            acc[t] = __builtin_amdgcn_wmma_f32_16x16x32_bf16(
                         false, a, false, b, (short)0, acc[t], false, false);
        }
    }

#pragma unroll
    for (int t = 0; t < 4; ++t) {
        const int ncol = ((ng << 2) + t) * 16 + (lane & 15);
        if (ncol < Nout) {                       // cold per-lane store guard
            const float bn = bias[ncol];
#pragma unroll
            for (int v = 0; v < 8; ++v) {
                const int mm = row0 + v + (half << 3);
                Y[(size_t)mm * Nout + ncol] = acc[t][v] + bn;
            }
        }
    }
}

// ---------------------------------------------------------------------------
// SpMM (COO, batched):  Z[b, rows[e], f] += vals[e] * Y[b, cols[e], f]
// ---------------------------------------------------------------------------
__global__ void spmm_atomic(const int* __restrict__ rows,
                            const int* __restrict__ cols,
                            const float* __restrict__ vals, int nnz,
                            const float* __restrict__ Y,
                            float* __restrict__ Z,
                            int Bn, int Nn, int F)
{
    const int e = blockIdx.x;
    if (e >= nnz) return;
    const int t = blockIdx.y * blockDim.x + threadIdx.x;
    if (t >= Bn * F) return;
    const int b = t / F;
    const int f = t - b * F;

    const int   r = rows[e];
    const int   c = cols[e];
    const float v = vals[e];

    const float g = v * Y[((size_t)b * Nn + c) * F + f];
    atomicAdd(&Z[((size_t)b * Nn + r) * F + f], g);
}

__global__ void fill_zero(float* __restrict__ p, size_t n)
{
    size_t i = (size_t)blockIdx.x * blockDim.x + threadIdx.x;
    if (i < n) p[i] = 0.0f;
}

__global__ void relu_inplace(float* __restrict__ p, size_t n)
{
    size_t i = (size_t)blockIdx.x * blockDim.x + threadIdx.x;
    if (i < n) p[i] = fmaxf(p[i], 0.0f);
}

// ---------------------------------------------------------------------------
// Host launcher
// ---------------------------------------------------------------------------
extern "C" void kernel_launch(void* const* d_in, const int* in_sizes, int n_in,
                              void* d_out, int out_size, void* d_ws, size_t ws_size,
                              hipStream_t stream)
{
    (void)n_in; (void)out_size; (void)ws_size;

    const float* H = (const float*)d_in[0];
    const int*   sm_s_r = (const int*)d_in[1];
    const int*   sm_s_c = (const int*)d_in[2];
    const float* sm_s_v = (const float*)d_in[3];
    const int*   sm_t_r = (const int*)d_in[4];
    const int*   sm_t_c = (const int*)d_in[5];
    const float* sm_t_v = (const float*)d_in[6];
    const int*   sp_s_r = (const int*)d_in[7];
    const int*   sp_s_c = (const int*)d_in[8];
    const float* sp_s_v = (const float*)d_in[9];
    const int*   sp_t_r = (const int*)d_in[10];
    const int*   sp_t_c = (const int*)d_in[11];
    const float* sp_t_v = (const float*)d_in[12];

    const float* Wmat[6] = { (const float*)d_in[13], (const float*)d_in[15],
                             (const float*)d_in[17], (const float*)d_in[19],
                             (const float*)d_in[21], (const float*)d_in[23] };
    const float* Bias[6] = { (const float*)d_in[14], (const float*)d_in[16],
                             (const float*)d_in[18], (const float*)d_in[20],
                             (const float*)d_in[22], (const float*)d_in[24] };
    const int Kdim[6] = {   2, 400, 300, 100, 300, 400 };
    const int Ndim[6] = { 400, 300, 100, 300, 400,   2 };

    const int nnz = in_sizes[1];
    const int Bn = 256, Nn = 255;
    const int M  = Bn * Nn;                      // 65280 (multiple of 16)

    float* bufA = (float*)d_ws;                  // 65280*400 f32
    float* bufB = bufA + (size_t)M * 400;
    __bf16* wpBase = (__bf16*)(bufB + (size_t)M * 400);
    float* out = (float*)d_out;

    // ---- pack weights into grouped WMMA B-fragment layout (per launch) ----
    __bf16* Wp[6];
    int ntg[6], kst[6];
    {
        size_t off = 0;
        for (int l = 0; l < 6; ++l) {
            const int ntn = (Ndim[l] + 15) / 16;
            ntg[l] = (ntn + 3) / 4;
            kst[l] = (Kdim[l] + 31) / 32;
            const int ntn_pad = ntg[l] * 4;
            Wp[l]  = wpBase + off;
            off   += (size_t)ntn_pad * kst[l] * 32 * 16;
            const int lanes = ntn_pad * kst[l] * 32;
            pack_w<<<(lanes + 255) / 256, 256, 0, stream>>>(
                Wmat[l], Wp[l], Kdim[l], Ndim[l], ntn_pad, kst[l]);
        }
    }

    auto gemm = [&](int l, const float* X, float* Y, bool relu_in) {
        const int tiles = (M / 16) * ntg[l];
        const int wpb   = 8;                     // 8 waves = 256 threads
        const int grid  = (tiles + wpb - 1) / wpb;
        if (relu_in)
            gemm_bias_wmma<true><<<grid, wpb * 32, 0, stream>>>(
                X, Wp[l], Bias[l], Y, M, Kdim[l], Ndim[l], ntg[l], kst[l]);
        else
            gemm_bias_wmma<false><<<grid, wpb * 32, 0, stream>>>(
                X, Wp[l], Bias[l], Y, M, Kdim[l], Ndim[l], ntg[l], kst[l]);
    };
    auto spmm = [&](const int* r, const int* c, const float* v,
                    const float* Y, float* Z, int F) {
        const size_t n = (size_t)M * F;
        fill_zero<<<(unsigned)((n + 255) / 256), 256, 0, stream>>>(Z, n);
        dim3 grid((unsigned)nnz, (unsigned)((Bn * F + 255) / 256));
        spmm_atomic<<<grid, 256, 0, stream>>>(r, c, v, nnz, Y, Z, Bn, Nn, F);
    };

    // ---- encoder ----
    gemm(0, H,    bufA, false);  spmm(sm_s_r, sm_s_c, sm_s_v, bufA, bufB, 400);
    gemm(1, bufB, bufA, true);   spmm(sm_s_r, sm_s_c, sm_s_v, bufA, bufB, 300);
    gemm(2, bufB, bufA, true);   spmm(sm_t_r, sm_t_c, sm_t_v, bufA, bufB, 100);
    // ---- decoder ----
    gemm(3, bufB, bufA, true);   spmm(sp_t_r, sp_t_c, sp_t_v, bufA, bufB, 300);
    gemm(4, bufB, bufA, true);   spmm(sp_s_r, sp_s_c, sp_s_v, bufA, bufB, 400);
    gemm(5, bufB, bufA, true);   spmm(sp_s_r, sp_s_c, sp_s_v, bufA, out,    2);

    const size_t nout = (size_t)M * 2;
    relu_inplace<<<(unsigned)((nout + 255) / 256), 256, 0, stream>>>(out, nout);
}
